// Transformer_20220706030423
// MI455X (gfx1250) — matrix operand order repaired and checked by reading, not compiled, not run on hardware
//
#include <hip/hip_runtime.h>
#include <cstddef>

// ---------------- problem constants (match reference) ----------------
constexpr int V_  = 32000;
constexpr int E_  = 1024;
constexpr int S_  = 1024;
constexpr int B_  = 4;
constexpr int NL_ = 4;
constexpr int H_  = 16;
constexpr int DH_ = 64;
constexpr int DFF_= 4096;
constexpr int M_  = B_ * S_;           // 4096 rows in every GEMM

typedef unsigned short u16;
typedef unsigned int   u32;
typedef unsigned long long u64;
typedef __attribute__((ext_vector_type(16))) __bf16 bf16x16;
typedef __attribute__((ext_vector_type(2)))  __bf16 bf16x2;
typedef __attribute__((ext_vector_type(8)))  float  f32x8;
typedef __attribute__((ext_vector_type(4)))  float  f32x4;
typedef __attribute__((ext_vector_type(4)))  u32    u32x4;

union Frag {            // 32 bytes: raw halves <-> WMMA bf16 operand
  u32x4   u4[2];
  bf16x16 v;
};

// native bf16 converts -> backend can pick v_cvt_pk_bf16_f32
__device__ __forceinline__ u16 f2bf(float f) {
  __bf16 h = (__bf16)f;
  return __builtin_bit_cast(u16, h);
}
__device__ __forceinline__ u32 f2bf2(float a, float b) {
  bf16x2 p; p[0] = (__bf16)a; p[1] = (__bf16)b;
  return __builtin_bit_cast(u32, p);
}

__device__ __forceinline__ f32x8 vzero8() {
  f32x8 z = {0.f,0.f,0.f,0.f,0.f,0.f,0.f,0.f};
  return z;
}

// CDNA5 async DMA: global -> LDS, 16B per lane (+offset:16 second half),
// tracked by ASYNCcnt. ldsb = LDS byte address (low 32 bits of flat shared ptr).
__device__ __forceinline__ void async_load_lds_b128x2(u32 ldsb, const void* gsrc) {
  asm volatile("global_load_async_to_lds_b128 %0, %1, off"
               :: "v"(ldsb), "v"(gsrc) : "memory");
  asm volatile("global_load_async_to_lds_b128 %0, %1, off offset:16"
               :: "v"(ldsb), "v"(gsrc) : "memory");
}
__device__ __forceinline__ void wait_asynccnt0() {
  asm volatile("s_wait_asynccnt 0" ::: "memory");
}

// ---------------- weight fp32 [K][N] -> bf16 transposed [N][K] ----------------
__global__ __launch_bounds__(256) void convt_kernel(const float* __restrict__ W,
                                                    u16* __restrict__ Wt,
                                                    int K, int N) {
  size_t idx = (size_t)blockIdx.x * 256 + threadIdx.x;   // over K*N
  size_t k = idx / (size_t)N, n = idx % (size_t)N;
  Wt[n * (size_t)K + k] = f2bf(W[idx]);
}

// ---------------- embedding + positional encoding ----------------
__global__ __launch_bounds__(256) void embed_kernel(const int* __restrict__ tokens,
                                                    const float* __restrict__ emb,
                                                    const float* __restrict__ pe,
                                                    float* __restrict__ x) {
  const int row = blockIdx.x;           // b*S + s
  const int s   = row % S_;
  const int tok = tokens[row];
  const float* er = emb + (size_t)tok * E_;
  const float* pr = pe  + (size_t)s   * E_;
  float* xr = x + (size_t)row * E_;
  #pragma unroll
  for (int i = 0; i < 4; ++i) {
    int e = threadIdx.x + i * 256;
    xr[e] = er[e] + pr[e];
  }
}

// ---------------- block reduction helper ----------------
__device__ __forceinline__ float block_reduce_sum(float v, float* sred, int tid) {
  sred[tid] = v; __syncthreads();
  #pragma unroll
  for (int o = 128; o > 0; o >>= 1) {
    if (tid < o) sred[tid] += sred[tid + o];
    __syncthreads();
  }
  float r = sred[0]; __syncthreads();
  return r;
}

// x[row] += LN(a[row]) * g + b
__global__ __launch_bounds__(256) void ln_add_kernel(float* __restrict__ x,
                                                     const float* __restrict__ a,
                                                     const float* __restrict__ g,
                                                     const float* __restrict__ b) {
  __shared__ float sred[256];
  const int row = blockIdx.x, tid = threadIdx.x;
  const float* ar = a + (size_t)row * E_;
  float loc[4]; float s = 0.f;
  #pragma unroll
  for (int i = 0; i < 4; ++i) { loc[i] = ar[tid + i * 256]; s += loc[i]; }
  float mu = block_reduce_sum(s, sred, tid) * (1.f / E_);
  float vs = 0.f;
  #pragma unroll
  for (int i = 0; i < 4; ++i) { float d = loc[i] - mu; vs += d * d; }
  float var = block_reduce_sum(vs, sred, tid) * (1.f / E_);
  float inv = rsqrtf(var + 1e-6f);
  float* xr = x + (size_t)row * E_;
  #pragma unroll
  for (int i = 0; i < 4; ++i) {
    int e = tid + i * 256;
    xr[e] += (loc[i] - mu) * inv * g[e] + b[e];
  }
}

// x[row] = LN(x[row]) * g + b  (block owns full row -> in-place safe)
__global__ __launch_bounds__(256) void ln_inplace_kernel(float* __restrict__ x,
                                                         const float* __restrict__ g,
                                                         const float* __restrict__ b) {
  __shared__ float sred[256];
  const int row = blockIdx.x, tid = threadIdx.x;
  float* xr = x + (size_t)row * E_;
  float loc[4]; float s = 0.f;
  #pragma unroll
  for (int i = 0; i < 4; ++i) { loc[i] = xr[tid + i * 256]; s += loc[i]; }
  float mu = block_reduce_sum(s, sred, tid) * (1.f / E_);
  float vs = 0.f;
  #pragma unroll
  for (int i = 0; i < 4; ++i) { float d = loc[i] - mu; vs += d * d; }
  float var = block_reduce_sum(vs, sred, tid) * (1.f / E_);
  float inv = rsqrtf(var + 1e-6f);
  #pragma unroll
  for (int i = 0; i < 4; ++i) {
    int e = tid + i * 256;
    xr[e] = (loc[i] - mu) * inv * g[e] + b[e];
  }
}

// ---------------- WMMA bf16 GEMM: C[M,N] = A[M,K] (f32) * Bt[N,K] (bf16) + bias ----------------
// block tile 128x128, 256 threads = 8 waves (2 M-waves x 4 N-waves), wave tile 64x32, K step 32.
// B tile staged via CDNA5 async global->LDS DMA (overlaps with A fp32->bf16 VALU conversion).
template<bool RELU, bool OUTB>
__global__ __launch_bounds__(256) void gemm_bf16_kernel(const float* __restrict__ A,
                                                        const u16* __restrict__ Bt,
                                                        const float* __restrict__ bias,
                                                        float* __restrict__ Cf,
                                                        u16* __restrict__ Cb,
                                                        int N, int K) {
  __shared__ u16 As[128 * 32];   // [m][k]  bf16
  __shared__ u16 Bs[128 * 32];   // [n][k]  bf16 (already transposed in global)
  const int m0 = blockIdx.x * 128, n0 = blockIdx.y * 128;
  const int tid = threadIdx.x, wave = tid >> 5, lane = tid & 31;
  const int wm = wave >> 2, wn = wave & 3;
  const int half = lane >> 4, l16 = lane & 15;

  f32x8 acc[4][2];
  #pragma unroll
  for (int mi = 0; mi < 4; ++mi)
    #pragma unroll
    for (int ni = 0; ni < 2; ++ni) acc[mi][ni] = vzero8();

  const int ra = tid >> 1, ka = (tid & 1) * 16;     // each thread: one 16-elem half-row
  const float* aptr = A  + (size_t)(m0 + ra) * K + ka;
  const u16*   bptr = Bt + (size_t)(n0 + ra) * K + ka;
  const u32 ldsb = (u32)(size_t)(&Bs[ra * 32 + ka]);  // LDS byte address of my slot

  for (int k0 = 0; k0 < K; k0 += 32) {
    // B tile: async DMA straight into LDS (ASYNCcnt)
    async_load_lds_b128x2(ldsb, (const void*)(bptr + k0));

    // A tile: fp32 -> bf16 (packed converts), overlaps with the DMA above
    const f32x4* a4 = (const f32x4*)(aptr + k0);
    #pragma unroll
    for (int i = 0; i < 4; ++i) {
      f32x4 f = a4[i];
      *(u32*)&As[ra * 32 + ka + i * 4]     = f2bf2(f[0], f[1]);
      *(u32*)&As[ra * 32 + ka + i * 4 + 2] = f2bf2(f[2], f[3]);
    }
    if (k0 + 32 < K) {               // global_prefetch_b8 for next A tile
      __builtin_prefetch(aptr + k0 + 32, 0, 1);
    }
    wait_asynccnt0();                // my async B stores landed in LDS
    __syncthreads();                 // everyone's tiles visible

    #pragma unroll
    for (int mi = 0; mi < 4; ++mi) {
      Frag a;
      const int arow = (wm * 64 + mi * 16 + l16) * 32;
      a.u4[0] = *(const u32x4*)&As[arow + half * 8];        // k = half*8 .. +7
      a.u4[1] = *(const u32x4*)&As[arow + 16 + half * 8];   // k = 16+half*8 .. +7
      #pragma unroll
      for (int ni = 0; ni < 2; ++ni) {
        Frag bq;
        const int brow = (wn * 32 + ni * 16 + l16) * 32;
        bq.u4[0] = *(const u32x4*)&Bs[brow + half * 8];      // k = half*8 .. +7
        bq.u4[1] = *(const u32x4*)&Bs[brow + 16 + half * 8];
        acc[mi][ni] = __builtin_amdgcn_wmma_f32_16x16x32_bf16(
            false, a.v, false, bq.v, (short)0, acc[mi][ni], false, false);
      }
    }
    __syncthreads();
  }

  // epilogue: bias (+ReLU), f32 or bf16 output
  #pragma unroll
  for (int mi = 0; mi < 4; ++mi) {
    #pragma unroll
    for (int ni = 0; ni < 2; ++ni) {
      const int n = n0 + wn * 32 + ni * 16 + l16;
      const float bsv = bias[n];
      #pragma unroll
      for (int j = 0; j < 8; ++j) {
        const int m = m0 + wm * 64 + mi * 16 + half * 8 + j;
        float vv = acc[mi][ni][j] + bsv;
        if (RELU) vv = fmaxf(vv, 0.f);
        if (OUTB) Cb[(size_t)m * N + n] = f2bf(vv);
        else      Cf[(size_t)m * N + n] = vv;
      }
    }
  }
}

// ---------------- flash attention, WMMA for QK^T and PV ----------------
// grid (S/64, H, B), 128 threads = 4 waves; wave w owns query rows qb*64 + w*16 .. +15.
__global__ __launch_bounds__(128) void flash_attn_kernel(const u16* __restrict__ q,
                                                         const u16* __restrict__ k,
                                                         const u16* __restrict__ v,
                                                         float* __restrict__ out) {
  __shared__ u16 Pl[4][16 * 64];   // per-wave P tile (row-major [16][64] bf16)
  __shared__ u16 Vt[64 * 64];      // transposed V tile: [dim][key] bf16
  const int qb = blockIdx.x, h = blockIdx.y, b = blockIdx.z;
  const int tid = threadIdx.x, wave = tid >> 5, lane = tid & 31;
  const int half = lane >> 4, l16 = lane & 15;

  // Q fragments, hoisted: row = l16 (A-layout), two K-steps over DH=64
  const int qrow = qb * 64 + wave * 16 + l16;
  const u16* qp = q + ((size_t)(b * S_ + qrow)) * E_ + h * DH_;
  Frag aq[2];
  #pragma unroll
  for (int s = 0; s < 2; ++s) {
    aq[s].u4[0] = *(const u32x4*)(qp + s * 32 + half * 8);
    aq[s].u4[1] = *(const u32x4*)(qp + s * 32 + 16 + half * 8);
  }

  f32x8 o[4]; f32x8 mrun, lrun;
  #pragma unroll
  for (int j = 0; j < 4; ++j) o[j] = vzero8();
  #pragma unroll
  for (int e = 0; e < 8; ++e) { mrun[e] = -3e38f; lrun[e] = 0.f; }

  for (int t = 0; t < S_ / 64; ++t) {
    __syncthreads();                     // all waves done reading previous Vt
    {                                    // stage V tile transposed: Vt[dim][key]
      const int key = t * 64 + (tid >> 1);
      const int nb  = (tid & 1) * 32;
      const u16* vp = v + ((size_t)(b * S_ + key)) * E_ + h * DH_ + nb;
      const u32x4* vp4 = (const u32x4*)vp;          // 4 x b128 = 32 bf16
      #pragma unroll
      for (int i = 0; i < 4; ++i) {
        u32x4 r = vp4[i];
        #pragma unroll
        for (int c = 0; c < 4; ++c) {
          const int d = nb + (i * 4 + c) * 2;
          Vt[(d    ) * 64 + (tid >> 1)] = (u16)(r[c] & 0xFFFFu);
          Vt[(d + 1) * 64 + (tid >> 1)] = (u16)(r[c] >> 16);
        }
      }
    }
    __syncthreads();

    // S = Q * K^T (16x64 per wave): B[kdim][nkey] = K[key][kdim] -> contiguous K rows
    f32x8 sc[4];
    #pragma unroll
    for (int j = 0; j < 4; ++j) sc[j] = vzero8();
    #pragma unroll
    for (int s = 0; s < 2; ++s) {
      #pragma unroll
      for (int j = 0; j < 4; ++j) {
        const int key = t * 64 + j * 16 + l16;
        const u16* kp = k + ((size_t)(b * S_ + key)) * E_ + h * DH_ + s * 32 + half * 8;
        Frag bk;
        bk.u4[0] = *(const u32x4*)kp;
        bk.u4[1] = *(const u32x4*)(kp + 16);
        sc[j] = __builtin_amdgcn_wmma_f32_16x16x32_bf16(
            false, aq[s].v, false, bk.v, (short)0, sc[j], false, false);
      }
    }

    // scale + row max (row = e + half*8; columns striped over 16-lane halves)
    f32x8 mx;
    #pragma unroll
    for (int e = 0; e < 8; ++e) {
      float a0 = sc[0][e] * 0.125f, a1 = sc[1][e] * 0.125f;
      float a2 = sc[2][e] * 0.125f, a3 = sc[3][e] * 0.125f;
      sc[0][e] = a0; sc[1][e] = a1; sc[2][e] = a2; sc[3][e] = a3;
      mx[e] = fmaxf(fmaxf(a0, a1), fmaxf(a2, a3));
    }
    #pragma unroll
    for (int off = 1; off < 16; off <<= 1)
      #pragma unroll
      for (int e = 0; e < 8; ++e) mx[e] = fmaxf(mx[e], __shfl_xor(mx[e], off, 32));

    // online softmax update
    f32x8 mnew, corr, psum;
    #pragma unroll
    for (int e = 0; e < 8; ++e) {
      mnew[e] = fmaxf(mrun[e], mx[e]);
      corr[e] = __expf(mrun[e] - mnew[e]);
      psum[e] = 0.f;
    }
    #pragma unroll
    for (int j = 0; j < 4; ++j)
      #pragma unroll
      for (int e = 0; e < 8; ++e) {
        float p = __expf(sc[j][e] - mnew[e]);
        sc[j][e] = p; psum[e] += p;
      }
    #pragma unroll
    for (int off = 1; off < 16; off <<= 1)
      #pragma unroll
      for (int e = 0; e < 8; ++e) psum[e] += __shfl_xor(psum[e], off, 32);
    #pragma unroll
    for (int e = 0; e < 8; ++e) {
      lrun[e] = lrun[e] * corr[e] + psum[e];
      mrun[e] = mnew[e];
    }
    #pragma unroll
    for (int j = 0; j < 4; ++j)
      #pragma unroll
      for (int e = 0; e < 8; ++e) o[j][e] *= corr[e];

    // repack P (C-layout) -> LDS row-major [16][64] (wave-local; DS ops in-order)
    u16* pw = Pl[wave];
    #pragma unroll
    for (int j = 0; j < 4; ++j)
      #pragma unroll
      for (int e = 0; e < 8; ++e)
        pw[(half * 8 + e) * 64 + j * 16 + l16] = f2bf(sc[j][e]);

    // P A-fragments (two K-steps over 64 keys)
    Frag ap[2];
    #pragma unroll
    for (int s = 0; s < 2; ++s) {
      ap[s].u4[0] = *(const u32x4*)&pw[l16 * 64 + s * 32 + half * 8];
      ap[s].u4[1] = *(const u32x4*)&pw[l16 * 64 + s * 32 + 16 + half * 8];
    }
    // O += P * V : B[key][dim] read from transposed Vt[dim][key] (contiguous keys)
    #pragma unroll
    for (int s = 0; s < 2; ++s) {
      #pragma unroll
      for (int j = 0; j < 4; ++j) {
        Frag bv;
        const u16* vt = &Vt[(j * 16 + l16) * 64 + s * 32 + half * 8];
        bv.u4[0] = *(const u32x4*)vt;
        bv.u4[1] = *(const u32x4*)(vt + 16);
        o[j] = __builtin_amdgcn_wmma_f32_16x16x32_bf16(
            false, ap[s].v, false, bv.v, (short)0, o[j], false, false);
      }
    }
  }

  // normalize and write [B,S,E] fp32 (heads concatenated)
  #pragma unroll
  for (int j = 0; j < 4; ++j)
    #pragma unroll
    for (int e = 0; e < 8; ++e) {
      const int row = qb * 64 + wave * 16 + half * 8 + e;
      const int col = h * DH_ + j * 16 + l16;
      out[((size_t)(b * S_ + row)) * E_ + col] = o[j][e] / lrun[e];
    }
}

// ---------------- host orchestration ----------------
extern "C" void kernel_launch(void* const* d_in, const int* in_sizes, int n_in,
                              void* d_out, int out_size, void* d_ws, size_t ws_size,
                              hipStream_t stream) {
  (void)in_sizes; (void)n_in; (void)out_size; (void)ws_size;

  const int*   tokens = (const int*)  d_in[0];
  const float* emb    = (const float*)d_in[1];
  const float* pe     = (const float*)d_in[2];
  const float* Wq     = (const float*)d_in[3];
  const float* Wk     = (const float*)d_in[4];
  const float* Wv     = (const float*)d_in[5];
  const float* Wo     = (const float*)d_in[6];
  const float* W1     = (const float*)d_in[7];
  const float* W2     = (const float*)d_in[8];
  const float* bq     = (const float*)d_in[9];
  const float* bk     = (const float*)d_in[10];
  const float* bv     = (const float*)d_in[11];
  const float* bo     = (const float*)d_in[12];
  const float* b1     = (const float*)d_in[13];
  const float* b2     = (const float*)d_in[14];
  const float* ln1g   = (const float*)d_in[15];
  const float* ln1b   = (const float*)d_in[16];
  const float* ln2g   = (const float*)d_in[17];
  const float* ln2b   = (const float*)d_in[18];
  const float* encg   = (const float*)d_in[19];
  const float* encb   = (const float*)d_in[20];
  const float* fing   = (const float*)d_in[21];
  const float* finb   = (const float*)d_in[22];
  const float* Wout   = (const float*)d_in[23];
  const float* bout   = (const float*)d_in[24];

  const size_t EE = (size_t)E_ * E_;
  const size_t ED = (size_t)E_ * DFF_;
  const size_t EV = (size_t)E_ * V_;
  const size_t ME = (size_t)M_ * E_;
  const size_t MD = (size_t)M_ * DFF_;

  char* wp = (char*)d_ws;
  auto carve = [&](size_t bytes) -> void* {
    void* p = (void*)wp;
    wp += (bytes + 255) & ~(size_t)255;
    return p;
  };
  u16* wqT   = (u16*)carve(NL_ * EE * 2);
  u16* wkT   = (u16*)carve(NL_ * EE * 2);
  u16* wvT   = (u16*)carve(NL_ * EE * 2);
  u16* woT   = (u16*)carve(NL_ * EE * 2);
  u16* w1T   = (u16*)carve(NL_ * ED * 2);
  u16* w2T   = (u16*)carve(NL_ * ED * 2);
  u16* woutT = (u16*)carve(EV * 2);
  u16* qbuf  = (u16*)carve(ME * 2);
  u16* kbuf  = (u16*)carve(ME * 2);
  u16* vbuf  = (u16*)carve(ME * 2);
  float* x    = (float*)carve(ME * 4);
  float* tmp  = (float*)carve(ME * 4);
  float* att  = (float*)carve(ME * 4);
  float* hbuf = (float*)carve(MD * 4);

  // 1) convert + transpose all weights to bf16 [N][K]
  for (int i = 0; i < NL_; ++i) {
    convt_kernel<<<(int)(EE / 256), 256, 0, stream>>>(Wq + i * EE, wqT + i * EE, E_, E_);
    convt_kernel<<<(int)(EE / 256), 256, 0, stream>>>(Wk + i * EE, wkT + i * EE, E_, E_);
    convt_kernel<<<(int)(EE / 256), 256, 0, stream>>>(Wv + i * EE, wvT + i * EE, E_, E_);
    convt_kernel<<<(int)(EE / 256), 256, 0, stream>>>(Wo + i * EE, woT + i * EE, E_, E_);
    convt_kernel<<<(int)(ED / 256), 256, 0, stream>>>(W1 + i * ED, w1T + i * ED, E_, DFF_);
    convt_kernel<<<(int)(ED / 256), 256, 0, stream>>>(W2 + i * ED, w2T + i * ED, DFF_, E_);
  }
  convt_kernel<<<(int)(EV / 256), 256, 0, stream>>>(Wout, woutT, E_, V_);

  // 2) embedding + positional encoding
  embed_kernel<<<M_, 256, 0, stream>>>(tokens, emb, pe, x);

  // 3) encoder layers
  const dim3 gE(M_ / 128, E_ / 128);     // (32, 8)
  const dim3 gF(M_ / 128, DFF_ / 128);   // (32, 32)
  for (int i = 0; i < NL_; ++i) {
    gemm_bf16_kernel<false, true><<<gE, 256, 0, stream>>>(x, wqT + i * EE, bq + i * E_, nullptr, qbuf, E_, E_);
    gemm_bf16_kernel<false, true><<<gE, 256, 0, stream>>>(x, wkT + i * EE, bk + i * E_, nullptr, kbuf, E_, E_);
    gemm_bf16_kernel<false, true><<<gE, 256, 0, stream>>>(x, wvT + i * EE, bv + i * E_, nullptr, vbuf, E_, E_);
    flash_attn_kernel<<<dim3(S_ / 64, H_, B_), 128, 0, stream>>>(qbuf, kbuf, vbuf, att);
    gemm_bf16_kernel<false, false><<<gE, 256, 0, stream>>>(att, woT + i * EE, bo + i * E_, tmp, nullptr, E_, E_);
    ln_add_kernel<<<M_, 256, 0, stream>>>(x, tmp, ln1g + i * E_, ln1b + i * E_);
    gemm_bf16_kernel<true, false><<<gF, 256, 0, stream>>>(x, w1T + i * ED, b1 + i * DFF_, hbuf, nullptr, DFF_, E_);
    gemm_bf16_kernel<false, false><<<gE, 256, 0, stream>>>(hbuf, w2T + i * ED, b2 + i * E_, tmp, nullptr, E_, DFF_);
    ln_add_kernel<<<M_, 256, 0, stream>>>(x, tmp, ln2g + i * E_, ln2b + i * E_);
  }

  // 4) final LNs + logits
  ln_inplace_kernel<<<M_, 256, 0, stream>>>(x, encg, encb);
  ln_inplace_kernel<<<M_, 256, 0, stream>>>(x, fing, finb);
  gemm_bf16_kernel<false, false><<<dim3(M_ / 128, V_ / 128), 256, 0, stream>>>(
      x, woutT, bout, (float*)d_out, nullptr, V_, E_);
}
